// GRUMappingGCN0_42073499631908
// MI455X (gfx1250) — matrix operand order repaired and checked by reading, not compiled, not run on hardware
//
#include <hip/hip_runtime.h>

typedef __bf16 bf16;
typedef __attribute__((ext_vector_type(16))) __bf16 v16bf;
typedef __attribute__((ext_vector_type(8)))  __bf16 v8bf;
typedef __attribute__((ext_vector_type(8)))  float  v8f;
typedef __attribute__((ext_vector_type(4)))  float  v4f;

#define LEAKY_SLOPE 0.1f
#define NB   16      // B
#define NN1  128     // N1
#define NN2  8192    // N2
#define NC   256     // channels
#define NK   64      // nsample
#define SFULL (NB*NK*NN1)   // 131072
#define S1    (NB*NN1)      // 2048
#define CIN0  288           // 259 padded to multiple of 32

// ---------------------------------------------------------------------------
// points2 [B,256,8192] fp32  ->  p2t [B,8192,256] bf16  (tiled LDS transpose)
// ---------------------------------------------------------------------------
__global__ void transpose_cvt_kernel(const float* __restrict__ src,
                                     bf16* __restrict__ dst) {
  __shared__ float tile[32][33];
  int j0 = blockIdx.x * 32;
  int c0 = blockIdx.y * 32;
  int b  = blockIdx.z;
  int tx = threadIdx.x & 31;
  int ty = threadIdx.x >> 5;           // 0..7
  for (int r = ty; r < 32; r += 8)
    tile[r][tx] = src[((size_t)(b * NC + c0 + r)) * NN2 + j0 + tx];
  __syncthreads();
  for (int r = ty; r < 32; r += 8)
    dst[((size_t)(b * NN2 + j0 + r)) * NC + c0 + tx] = (bf16)tile[tx][r];
}

// ---------------------------------------------------------------------------
// KNN: one block per (b,n1); distances in LDS, 64x iterative argmin.
// K-order is irrelevant downstream (K-permutation invariant pipeline).
// ---------------------------------------------------------------------------
__global__ void knn_kernel(const float* __restrict__ xyz1,
                           const float* __restrict__ xyz2,
                           int* __restrict__ knn_idx) {
  __shared__ float sd[NN2];
  __shared__ float red[256];
  __shared__ int   redi[256];
  int bn = blockIdx.x;                 // b*128 + n1
  int b  = bn >> 7;
  int n1 = bn & 127;
  int tid = threadIdx.x;
  float qx = xyz1[(b * 3 + 0) * NN1 + n1];
  float qy = xyz1[(b * 3 + 1) * NN1 + n1];
  float qz = xyz1[(b * 3 + 2) * NN1 + n1];
  const float* x2b = xyz2 + (size_t)b * 3 * NN2;
  for (int j = tid; j < NN2; j += 256) {
    float dx = x2b[j] - qx;
    float dy = x2b[NN2 + j] - qy;
    float dz = x2b[2 * NN2 + j] - qz;
    sd[j] = dx * dx + dy * dy + dz * dz;
  }
  __syncthreads();
  for (int t = 0; t < NK; ++t) {
    float best = 3.4e38f;
    int bi = 0;
    for (int j = tid; j < NN2; j += 256) {
      float v = sd[j];
      if (v < best) { best = v; bi = j; }
    }
    red[tid] = best; redi[tid] = bi;
    __syncthreads();
    for (int off = 128; off > 0; off >>= 1) {
      if (tid < off && red[tid + off] < red[tid]) {
        red[tid] = red[tid + off]; redi[tid] = redi[tid + off];
      }
      __syncthreads();
    }
    if (tid == 0) {
      knn_idx[bn * NK + t] = redi[0];
      sd[redi[0]] = 3.4e38f;           // remove winner
    }
    __syncthreads();
  }
}

// ---------------------------------------------------------------------------
// Build X0 bf16 [S, 288]:  c<256 = gathered features, 256..258 = dxyz, pad 0.
// s = ((b*64 + k)*128 + n1).  One wave per row.
// ---------------------------------------------------------------------------
__global__ void gather_pack_kernel(const bf16* __restrict__ p2t,
                                   const int* __restrict__ knn_idx,
                                   const float* __restrict__ xyz1,
                                   const float* __restrict__ xyz2,
                                   bf16* __restrict__ X0) {
  int s    = blockIdx.x * 8 + (threadIdx.x >> 5);
  int lane = threadIdx.x & 31;
  int n1 = s & 127;
  int bk = s >> 7;
  int k  = bk & 63;
  int b  = bk >> 6;
  int idx = knn_idx[((b * NN1 + n1) * NK) + k];
  const bf16* srow = p2t + ((size_t)(b * NN2 + idx)) * NC;
  bf16* drow = X0 + (size_t)s * CIN0;
  *(v8bf*)(drow + lane * 8) = *(const v8bf*)(srow + lane * 8);
  float val = 0.f;
  if (lane < 3)
    val = xyz2[(b * 3 + lane) * NN2 + idx] - xyz1[(b * 3 + lane) * NN1 + n1];
  drow[NC + lane] = (bf16)val;         // covers 256..287 exactly
}

// ---------------------------------------------------------------------------
// Weight fp32 -> bf16 with optional column zero-pad
// ---------------------------------------------------------------------------
__global__ void cvt_pad_kernel(const float* __restrict__ src,
                               bf16* __restrict__ dst,
                               int rows, int scols, int dcols) {
  int t = blockIdx.x * 256 + threadIdx.x;
  if (t >= rows * dcols) return;
  int r = t / dcols, c = t - r * dcols;
  dst[t] = (bf16)((c < scols) ? src[r * scols + c] : 0.f);
}

// ---------------------------------------------------------------------------
// pg[b,c,n1] = sum_k points1[b,c,k] * graph_a[c,k,n1]  -> bf16 [b*128+n1, c]
// block per (b,c), 128 threads (thread = n1); coalesced graph_a reads.
// ---------------------------------------------------------------------------
__global__ void pg_kernel(const float* __restrict__ points1,
                          const float* __restrict__ graph_a,
                          bf16* __restrict__ pg_in) {
  __shared__ float sp[NN1];
  int bc = blockIdx.x;
  int b = bc >> 8;
  int c = bc & 255;
  int n = threadIdx.x;
  sp[n] = points1[((size_t)(b * NC + c)) * NN1 + n];
  __syncthreads();
  const float* ga = graph_a + (size_t)c * NN1 * NN1;
  float acc = 0.f;
  for (int k = 0; k < NN1; ++k)
    acc = fmaf(sp[k], ga[k * NN1 + n], acc);
  pg_in[((size_t)(b * NN1 + n)) * NC + c] = (bf16)acc;
}

// ---------------------------------------------------------------------------
// WMMA GEMM:  out[s, m] = act( W[m,:]·X[s,:] + bias[m] + add ) (* mul)
// X: bf16 [S, Cin] row-major, W: bf16 [256, Cin] row-major, Cout = 256.
// Block = 256 thr (8 waves); wave -> 16(m) x 64(s) tile, 4 accumulators.
// add_mode: 0 none, 1 addbuf[s,m], 2 addbuf[fuse(s),m] (fuse = b*128+n1)
// mul_mode: 0 none, 2 mulbuf[fuse(s),m]
// act: 0 none, 1 leaky(0.1), 2 sigmoid
// ---------------------------------------------------------------------------
__global__ __launch_bounds__(256)
void wmma_gemm_kernel(const bf16* __restrict__ X, const bf16* __restrict__ W,
                      const float* __restrict__ bias,
                      const float* __restrict__ addbuf, int add_mode,
                      const float* __restrict__ mulbuf, int mul_mode,
                      int act,
                      bf16* __restrict__ out_bf, float* __restrict__ out_f32,
                      int S, int Cin) {
  int wave = threadIdx.x >> 5;
  int lane = threadIdx.x & 31;
  int mtile = blockIdx.y * 8 + wave;          // 0..15
  int sbase = blockIdx.x * 64;
  int half  = lane >> 4;                       // 0|1
  int ncol  = lane & 15;
  int a_koff = half * 8;                       // A frag k offset
  int b_koff = half * 8;                       // B frag k offset
  const bf16* wrow = W + (size_t)(mtile * 16 + ncol) * Cin;

  v8f z8 = {0.f, 0.f, 0.f, 0.f, 0.f, 0.f, 0.f, 0.f};
  v8f acc0 = z8, acc1 = z8, acc2 = z8, acc3 = z8;

  for (int k0 = 0; k0 < Cin; k0 += 32) {
    v16bf a;
    {
      v8bf lo = *(const v8bf*)(wrow + k0 + a_koff);
      v8bf hi = *(const v8bf*)(wrow + k0 + a_koff + 16);
#pragma unroll
      for (int i = 0; i < 8; ++i) { a[i] = lo[i]; a[i + 8] = hi[i]; }
    }
#pragma unroll
    for (int t = 0; t < 4; ++t) {
      const bf16* xrow = X + (size_t)(sbase + t * 16 + ncol) * Cin + k0 + b_koff;
      v16bf bf;
      v8bf lo = *(const v8bf*)(xrow);
      v8bf hi = *(const v8bf*)(xrow + 16);
#pragma unroll
      for (int i = 0; i < 8; ++i) { bf[i] = lo[i]; bf[i + 8] = hi[i]; }
      v8f& acc = (t == 0) ? acc0 : (t == 1) ? acc1 : (t == 2) ? acc2 : acc3;
      acc = __builtin_amdgcn_wmma_f32_16x16x32_bf16(
          false, a, false, bf, (short)0, acc, false, false);
    }
  }

  // Epilogue
  int mbase = mtile * 16 + half * 8;           // 8 contiguous output channels
  float bv[8];
#pragma unroll
  for (int r = 0; r < 8; ++r) bv[r] = bias ? bias[mbase + r] : 0.f;

#pragma unroll
  for (int t = 0; t < 4; ++t) {
    v8f& acc = (t == 0) ? acc0 : (t == 1) ? acc1 : (t == 2) ? acc2 : acc3;
    int s = sbase + t * 16 + ncol;
    int f = ((s >> 13) << 7) | (s & 127);      // b*128 + n1  (K*N1=2^13, N1=2^7)
    float res[8];
#pragma unroll
    for (int r = 0; r < 8; ++r) {
      float v = acc[r] + bv[r];
      if (add_mode == 1)      v += addbuf[(size_t)s * NC + mbase + r];
      else if (add_mode == 2) v += addbuf[(size_t)f * NC + mbase + r];
      if (act == 1)      v = (v >= 0.f) ? v : LEAKY_SLOPE * v;
      else if (act == 2) v = 1.f / (1.f + expf(-v));
      if (mul_mode == 2) v *= mulbuf[(size_t)f * NC + mbase + r];
      res[r] = v;
    }
    if (out_bf) {
      v8bf o;
#pragma unroll
      for (int r = 0; r < 8; ++r) o[r] = (bf16)res[r];
      *(v8bf*)(out_bf + (size_t)s * NC + mbase) = o;
    }
    if (out_f32) {
      v4f lo = {res[0], res[1], res[2], res[3]};
      v4f hi = {res[4], res[5], res[6], res[7]};
      float* p = out_f32 + (size_t)s * NC + mbase;
      *(v4f*)p = lo;
      *(v4f*)(p + 4) = hi;
    }
  }
}

// ---------------------------------------------------------------------------
// max over K axis:  in bf16 [B*K*N1, 256] -> out bf16 [B*N1, 256]
// ---------------------------------------------------------------------------
__global__ void kmax_kernel(const bf16* __restrict__ in, bf16* __restrict__ out) {
  int t = blockIdx.x * 256 + threadIdx.x;      // 2048*256 threads
  int c  = t & 255;
  int s2 = t >> 8;                             // b*128 + n1
  int n1 = s2 & 127;
  int b  = s2 >> 7;
  size_t base = ((size_t)(b * NK) * NN1 + n1) * NC + c;
  float m = (float)in[base];
  for (int k = 1; k < NK; ++k)
    m = fmaxf(m, (float)in[base + (size_t)k * NN1 * NC]);
  out[t] = (bf16)m;
}

// ---------------------------------------------------------------------------
// out[b,c,n1] = (1-z)*pg + z*h     (all fp32 [b*128+n1, c] internally)
// ---------------------------------------------------------------------------
__global__ void combine_kernel(const float* __restrict__ zf,
                               const float* __restrict__ hf,
                               const float* __restrict__ pgf,
                               float* __restrict__ out) {
  int t = blockIdx.x * 256 + threadIdx.x;      // B*256*128 threads
  int n1 = t & 127;
  int c  = (t >> 7) & 255;
  int b  = t >> 15;
  int f = ((b * NN1 + n1) * NC) + c;
  float z = zf[f], h = hf[f], p = pgf[f];
  out[t] = (1.f - z) * p + z * h;
}

// ---------------------------------------------------------------------------
extern "C" void kernel_launch(void* const* d_in, const int* in_sizes, int n_in,
                              void* d_out, int out_size, void* d_ws, size_t ws_size,
                              hipStream_t stream) {
  const float* xyz1      = (const float*)d_in[0];
  const float* xyz2      = (const float*)d_in[1];
  const float* points1   = (const float*)d_in[2];
  const float* points2   = (const float*)d_in[3];
  const float* graph_a   = (const float*)d_in[4];
  const float* graph_w_w = (const float*)d_in[5];
  const float* graph_w_b = (const float*)d_in[6];
  const float* fuse_r_w  = (const float*)d_in[7];
  const float* fuse_z_w  = (const float*)d_in[8];
  const float* fuse_ro_w = (const float*)d_in[9];
  const float* r_w0 = (const float*)d_in[10]; const float* r_b0 = (const float*)d_in[11];
  const float* r_w1 = (const float*)d_in[12]; const float* r_b1 = (const float*)d_in[13];
  const float* r_w2 = (const float*)d_in[14]; const float* r_b2 = (const float*)d_in[15];
  const float* z_w0 = (const float*)d_in[16]; const float* z_b0 = (const float*)d_in[17];
  const float* z_w1 = (const float*)d_in[18]; const float* z_b1 = (const float*)d_in[19];
  const float* z_w2 = (const float*)d_in[20]; const float* z_b2 = (const float*)d_in[21];
  const float* h_w0 = (const float*)d_in[22]; const float* h_b0 = (const float*)d_in[23];
  const float* h_w1 = (const float*)d_in[24]; const float* h_b1 = (const float*)d_in[25];
  const float* h_w2 = (const float*)d_in[26]; const float* h_b2 = (const float*)d_in[27];

  char* ws = (char*)d_ws;
  size_t off = 0;
  auto take = [&](size_t bytes) -> char* {
    char* p = ws + off;
    off += (bytes + 255) & ~(size_t)255;
    return p;
  };

  int*  knn_idx = (int*)take((size_t)S1 * NK * 4);
  // E (fp32 [S,256], 134MB) aliases p2t (bf16 [B,8192,256], 67MB): p2t is dead
  // before E is written.
  char* e_region = take((size_t)SFULL * NC * 4);
  float* E   = (float*)e_region;
  bf16*  p2t = (bf16*)e_region;
  bf16*  X0  = (bf16*)take((size_t)SFULL * CIN0 * 2);
  bf16*  T1  = (bf16*)take((size_t)SFULL * NC * 2);
  bf16*  T2  = (bf16*)take((size_t)SFULL * NC * 2);
  bf16*  w0r = (bf16*)take((size_t)NC * CIN0 * 2);
  bf16*  w0z = (bf16*)take((size_t)NC * CIN0 * 2);
  bf16*  w0h = (bf16*)take((size_t)NC * CIN0 * 2);
  bf16*  w1r = (bf16*)take((size_t)NC * NC * 2);
  bf16*  w2r = (bf16*)take((size_t)NC * NC * 2);
  bf16*  w1z = (bf16*)take((size_t)NC * NC * 2);
  bf16*  w2z = (bf16*)take((size_t)NC * NC * 2);
  bf16*  w1h = (bf16*)take((size_t)NC * NC * 2);
  bf16*  w2h = (bf16*)take((size_t)NC * NC * 2);
  bf16*  gw  = (bf16*)take((size_t)NC * NC * 2);
  bf16*  frw = (bf16*)take((size_t)NC * NC * 2);
  bf16*  fzw = (bf16*)take((size_t)NC * NC * 2);
  bf16*  frow= (bf16*)take((size_t)NC * NC * 2);
  bf16*  pg_in = (bf16*)take((size_t)S1 * NC * 2);
  bf16*  pgb   = (bf16*)take((size_t)S1 * NC * 2);
  float* pgf   = (float*)take((size_t)S1 * NC * 4);
  float* Fr    = (float*)take((size_t)S1 * NC * 4);
  float* Fz    = (float*)take((size_t)S1 * NC * 4);
  bf16*  z1m   = (bf16*)take((size_t)S1 * NC * 2);
  bf16*  h1m   = (bf16*)take((size_t)S1 * NC * 2);
  float* zf    = (float*)take((size_t)S1 * NC * 4);
  float* hf    = (float*)take((size_t)S1 * NC * 4);

  // 1. transpose+convert points2 -> p2t
  {
    dim3 g(NN2 / 32, NC / 32, NB);
    transpose_cvt_kernel<<<g, 256, 0, stream>>>(points2, p2t);
  }
  // 2. knn
  knn_kernel<<<S1, 256, 0, stream>>>(xyz1, xyz2, knn_idx);
  // 3. gather + pack X0
  gather_pack_kernel<<<SFULL / 8, 256, 0, stream>>>(p2t, knn_idx, xyz1, xyz2, X0);
  // 4. weight conversions
  auto cvt = [&](const float* s, bf16* d, int rows, int sc, int dc) {
    int n = rows * dc;
    cvt_pad_kernel<<<(n + 255) / 256, 256, 0, stream>>>(s, d, rows, sc, dc);
  };
  cvt(r_w0, w0r, NC, 259, CIN0); cvt(z_w0, w0z, NC, 259, CIN0); cvt(h_w0, w0h, NC, 259, CIN0);
  cvt(r_w1, w1r, NC, NC, NC);    cvt(r_w2, w2r, NC, NC, NC);
  cvt(z_w1, w1z, NC, NC, NC);    cvt(z_w2, w2z, NC, NC, NC);
  cvt(h_w1, w1h, NC, NC, NC);    cvt(h_w2, w2h, NC, NC, NC);
  cvt(graph_w_w, gw, NC, NC, NC);
  cvt(fuse_r_w, frw, NC, NC, NC); cvt(fuse_z_w, fzw, NC, NC, NC);
  cvt(fuse_ro_w, frow, NC, NC, NC);
  // 5. graph mixing pg
  pg_kernel<<<NB * NC, NN1, 0, stream>>>(points1, graph_a, pg_in);

  auto gemm = [&](const bf16* X, const bf16* W, const float* bias,
                  const float* addbuf, int add_mode,
                  const float* mulbuf, int mul_mode, int act,
                  bf16* obf, float* of32, int S, int Cin) {
    dim3 g(S / 64, 2);
    wmma_gemm_kernel<<<g, 256, 0, stream>>>(X, W, bias, addbuf, add_mode,
                                            mulbuf, mul_mode, act, obf, of32,
                                            S, Cin);
  };

  // 6. point1_graph = leaky(gw @ pg + b) -> pgb (bf16) + pgf (f32)
  gemm(pg_in, gw, graph_w_b, nullptr, 0, nullptr, 0, 1, pgb, pgf, S1, NC);
  // 7/8. fuse terms
  gemm(pgb, frw, nullptr, nullptr, 0, nullptr, 0, 0, nullptr, Fr, S1, NC);
  gemm(pgb, fzw, nullptr, nullptr, 0, nullptr, 0, 0, nullptr, Fz, S1, NC);
  // 9-11. r gate
  gemm(X0, w0r, r_b0, Fr, 2, nullptr, 0, 1, T1, nullptr, SFULL, CIN0);   // r0
  gemm(T1, w1r, r_b1, nullptr, 0, nullptr, 0, 1, T2, nullptr, SFULL, NC); // r1
  gemm(T2, w2r, r_b2, nullptr, 0, pgf, 2, 2, T1, nullptr, SFULL, NC);     // Rp = sig(r2)*pg
  // 12. E = fuse_r_o_w @ Rp
  gemm(T1, frow, nullptr, nullptr, 0, nullptr, 0, 0, nullptr, E, SFULL, NC);
  // 13-16. z gate
  gemm(X0, w0z, z_b0, Fz, 2, nullptr, 0, 1, T2, nullptr, SFULL, CIN0);    // z0
  gemm(T2, w1z, z_b1, nullptr, 0, nullptr, 0, 1, T1, nullptr, SFULL, NC); // z1
  kmax_kernel<<<(S1 * NC) / 256, 256, 0, stream>>>(T1, z1m);
  gemm(z1m, w2z, z_b2, nullptr, 0, nullptr, 0, 2, nullptr, zf, S1, NC);   // z = sigmoid
  // 17-20. h gate
  gemm(X0, w0h, h_b0, E, 1, nullptr, 0, 1, T2, nullptr, SFULL, CIN0);     // h0
  gemm(T2, w1h, h_b1, nullptr, 0, nullptr, 0, 1, T1, nullptr, SFULL, NC); // h1
  kmax_kernel<<<(S1 * NC) / 256, 256, 0, stream>>>(T1, h1m);
  gemm(h1m, w2h, h_b2, nullptr, 0, nullptr, 0, 1, nullptr, hf, S1, NC);   // h = leaky
  // 21. combine -> d_out [B,256,128] fp32
  combine_kernel<<<(NB * NC * NN1) / 256, 256, 0, stream>>>(zf, hf, pgf, (float*)d_out);
}
